// ResidualBlock_80264348827724
// MI455X (gfx1250) — matrix operand order repaired and checked by reading, not compiled, not run on hardware
//
#include <hip/hip_runtime.h>
#include <hip/hip_bf16.h>

typedef __attribute__((ext_vector_type(16))) __bf16 v16bf;
typedef __attribute__((ext_vector_type(8)))  __bf16 v8bf;
typedef __attribute__((ext_vector_type(4)))  __bf16 v4bf;
typedef __attribute__((ext_vector_type(8)))  float  v8f;

#define BATCH   8
#define TLEN    16384
#define CRES    128
#define CDIL    256
#define CSKIP   256
#define TM      64          // timesteps per workgroup
#define XS_STRIDE 136       // 128 ch + 8 pad (68 dwords/row -> conflict-free)
#define MG_STRIDE 264       // 256 ch + 8 pad (132 dwords/row)

// ---------------------------------------------------------------------------
// Pack weights into WMMA-B-friendly bf16 layout in workspace:
//   Wg[n][kk]: n in [0,512) = output col (0..255 tanh, 256..511 sigmoid)
//              kk in [0,256): kk<128 -> tap0 (x[t-4]) cin=kk ; kk>=128 -> tap1 cin=kk-128
//   Wo[n][c] : n in [0,384) = output col (0..127 res, 128..383 skip), c = merged channel
// Row-major [n][k] so a lane's B fragment (16 consecutive K) is contiguous.
// ---------------------------------------------------------------------------
__global__ __launch_bounds__(256) void prep_weights(
    const float* __restrict__ w_tanh, const float* __restrict__ w_sig,
    const float* __restrict__ w_res,  const float* __restrict__ w_skip,
    __bf16* __restrict__ Wg, __bf16* __restrict__ Wo)
{
  int idx = blockIdx.x * 256 + threadIdx.x;
  const int NG = 512 * 256;
  const int NO = 384 * 256;
  if (idx < NG) {
    int n    = idx >> 8;        // 0..511
    int kk   = idx & 255;       // 0..255
    int tap  = kk >> 7;         // 0/1
    int cin  = kk & 127;
    int cout = n & 255;
    const float* src = (n < 256) ? w_tanh : w_sig;   // [2][128][256]
    Wg[n * 256 + kk] = (__bf16)src[tap * (128 * 256) + cin * 256 + cout];
  } else if (idx < NG + NO) {
    int j = idx - NG;
    int n = j >> 8;             // 0..383
    int c = j & 255;            // merged channel
    float v = (n < 128) ? w_res[c * 128 + n]          // [256][128]
                        : w_skip[c * 256 + (n - 128)];// [256][256]
    Wo[n * 256 + c] = (__bf16)v;
  }
}

// ---------------------------------------------------------------------------
// Main fused kernel: one workgroup = 64 timesteps of one batch row.
//   GEMM1: A=[x[t-4]||x[t]] (64x256) * Wg (256x512) -> gate, tanh*sigmoid -> LDS
//   GEMM2: merged (64x256) * Wo (256x384) -> res(+x+b) and skip(+b) outputs
// ---------------------------------------------------------------------------
__global__ __launch_bounds__(256) void wavenet_block(
    const float* __restrict__ x,
    const float* __restrict__ b_tanh, const float* __restrict__ b_sig,
    const float* __restrict__ b_res,  const float* __restrict__ b_skip,
    const __bf16* __restrict__ Wg, const __bf16* __restrict__ Wo,
    float* __restrict__ out)
{
  __shared__ __bf16 xs[(TM + 4) * XS_STRIDE];   // rows t0-4 .. t0+63, bf16
  __shared__ __bf16 mg[TM * MG_STRIDE];         // merged tile, bf16

  const int tid  = threadIdx.x;
  const int wave = tid >> 5;
  const int lane = tid & 31;
  const int half = lane >> 4;   // 0: lanes 0-15, 1: lanes 16-31
  const int l15  = lane & 15;

  const int b  = blockIdx.x >> 8;               // TLEN/TM = 256 blocks per batch
  const int t0 = (blockIdx.x & 255) * TM;

  // ---- Stage x tile (with 4-row causal halo) into LDS as bf16 ----
  for (int i = tid; i < (TM + 4) * (CRES / 4); i += 256) {
    int rr = i >> 5;            // row 0..67  (CRES/4 == 32 float4 per row)
    int c4 = (i & 31) * 4;
    int t  = t0 - 4 + rr;
    float4 v = make_float4(0.f, 0.f, 0.f, 0.f);
    if (t >= 0)
      v = *(const float4*)(x + ((size_t)b * TLEN + t) * CRES + c4);
    v4bf p;
    p[0] = (__bf16)v.x; p[1] = (__bf16)v.y; p[2] = (__bf16)v.z; p[3] = (__bf16)v.w;
    *(v4bf*)(&xs[rr * XS_STRIDE + c4]) = p;
  }
  __syncthreads();

  const v8f vzero = {0.f, 0.f, 0.f, 0.f, 0.f, 0.f, 0.f, 0.f};

  // ---- GEMM1: gate. Each wave handles 2 merged column tiles (tanh+sig pair).
  for (int p = 0; p < 2; ++p) {
    int nm   = wave + p * 8;          // merged N-tile 0..15
    int colg = nm * 16 + l15;         // gate column 0..255
    float bt = b_tanh[colg];
    float bs = b_sig[colg];
    const __bf16* wgT = Wg + (size_t)colg * 256;          // tanh rows
    const __bf16* wgS = Wg + (size_t)(256 + colg) * 256;  // sigmoid rows

    v8f accT[4], accS[4];
    for (int mi = 0; mi < 4; ++mi) { accT[mi] = vzero; accS[mi] = vzero; }

    for (int kc = 0; kc < 8; ++kc) {
      int kb = kc * 32 + half * 16;                 // B: 16 contiguous K per lane
      v16bf bT = *(const v16bf*)(wgT + kb);
      v16bf bS = *(const v16bf*)(wgS + kb);
      int rowAdd = (kc >= 4) ? 4 : 0;               // tap0 -> x[t-4], tap1 -> x[t]
      int cb = (kc & 3) * 32;
      for (int mi = 0; mi < 4; ++mi) {
        int r = mi * 16 + l15 + rowAdd;
        const __bf16* ap = &xs[r * XS_STRIDE + cb + half * 16];
        v8bf lo = *(const v8bf*)ap;                 // K = cb+{0..7} / {16..23}
        v8bf hi = *(const v8bf*)(ap + 8);           // K = cb+{8..15} / {24..31}
        v16bf a = __builtin_shufflevector(lo, hi, 0,1,2,3,4,5,6,7,8,9,10,11,12,13,14,15);
        accT[mi] = __builtin_amdgcn_wmma_f32_16x16x32_bf16(
            false, a, false, bT, (short)0, accT[mi], false, false);
        accS[mi] = __builtin_amdgcn_wmma_f32_16x16x32_bf16(
            false, a, false, bS, (short)0, accS[mi], false, false);
      }
    }

    // activations -> merged LDS tile
    for (int mi = 0; mi < 4; ++mi) {
      for (int i = 0; i < 8; ++i) {
        int row  = mi * 16 + i + half * 8;
        float tv = accT[mi][i] + bt;
        float sv = accS[mi][i] + bs;
        float e2 = __expf(-2.f * tv);
        float th = (1.f - e2) / (1.f + e2);
        float sg = 1.f / (1.f + __expf(-sv));
        mg[row * MG_STRIDE + colg] = (__bf16)(th * sg);
      }
    }
  }
  __syncthreads();

  // ---- GEMM2: merged @ Wo -> res/skip. 24 N-tiles / 8 waves = 3 each.
  const size_t skipBase = (size_t)BATCH * TLEN * CRES;
  for (int p = 0; p < 3; ++p) {
    int nt  = wave + p * 8;          // 0..23
    int col = nt * 16 + l15;         // 0..383
    float bias = (col < CRES) ? b_res[col] : b_skip[col - CRES];
    const __bf16* wo = Wo + (size_t)col * 256;

    v8f acc[4];
    for (int mi = 0; mi < 4; ++mi) acc[mi] = vzero;

    for (int kc = 0; kc < 8; ++kc) {
      int kb = kc * 32 + half * 16;
      v16bf bO = *(const v16bf*)(wo + kb);
      int cb = kc * 32;
      for (int mi = 0; mi < 4; ++mi) {
        int r = mi * 16 + l15;
        const __bf16* ap = &mg[r * MG_STRIDE + cb + half * 16];
        v8bf lo = *(const v8bf*)ap;
        v8bf hi = *(const v8bf*)(ap + 8);
        v16bf a = __builtin_shufflevector(lo, hi, 0,1,2,3,4,5,6,7,8,9,10,11,12,13,14,15);
        acc[mi] = __builtin_amdgcn_wmma_f32_16x16x32_bf16(
            false, a, false, bO, (short)0, acc[mi], false, false);
      }
    }

    for (int mi = 0; mi < 4; ++mi) {
      for (int i = 0; i < 8; ++i) {
        int row = mi * 16 + i + half * 8;
        int t   = t0 + row;
        float v = acc[mi][i] + bias;
        if (col < CRES) {
          size_t xoff = ((size_t)b * TLEN + t) * CRES + col;
          out[xoff] = v + x[xoff];                       // residual add
        } else {
          out[skipBase + ((size_t)b * TLEN + t) * CSKIP + (col - CRES)] = v;
        }
      }
    }
  }
}

// ---------------------------------------------------------------------------
extern "C" void kernel_launch(void* const* d_in, const int* in_sizes, int n_in,
                              void* d_out, int out_size, void* d_ws, size_t ws_size,
                              hipStream_t stream) {
  const float* x      = (const float*)d_in[0];
  const float* w_tanh = (const float*)d_in[1];
  const float* b_tanh = (const float*)d_in[2];
  const float* w_sig  = (const float*)d_in[3];
  const float* b_sig  = (const float*)d_in[4];
  const float* w_res  = (const float*)d_in[5];
  const float* b_res  = (const float*)d_in[6];
  const float* w_skip = (const float*)d_in[7];
  const float* b_skip = (const float*)d_in[8];

  __bf16* Wg = (__bf16*)d_ws;            // 512*256 bf16 = 256 KB
  __bf16* Wo = Wg + 512 * 256;           // 384*256 bf16 = 192 KB

  const int prepElems = 512 * 256 + 384 * 256;
  prep_weights<<<(prepElems + 255) / 256, 256, 0, stream>>>(
      w_tanh, w_sig, w_res, w_skip, Wg, Wo);

  const int nBlocks = (BATCH * TLEN) / TM;  // 2048
  wavenet_block<<<nBlocks, 256, 0, stream>>>(
      x, b_tanh, b_sig, b_res, b_skip, Wg, Wo, (float*)d_out);
}
